// CompositionalCritic_18116172054929
// MI455X (gfx1250) — matrix operand order repaired and checked by reading, not compiled, not run on hardware
//
#include <hip/hip_runtime.h>

// ---------- types ----------
typedef __bf16 bf16_t;
typedef __attribute__((ext_vector_type(16))) bf16_t          v16bf;
typedef __attribute__((ext_vector_type(8)))  float           v8f;
typedef __attribute__((ext_vector_type(8)))  unsigned short  v8u16;
typedef __attribute__((ext_vector_type(16))) unsigned short  v16u16;
typedef __attribute__((ext_vector_type(4)))  int             v4i;

#if __has_builtin(__builtin_amdgcn_global_load_async_to_lds_b128) && \
    __has_builtin(__builtin_amdgcn_s_wait_asynccnt)
#define USE_ASYNC 1
#else
#define USE_ASYNC 0
#endif

// async-to-LDS builtin: (global v4i* src, local v4i* dst, imm offset, imm cpol)
#define ASG(p) ((__attribute__((address_space(1))) v4i*)(p))
#define ASL(p) ((__attribute__((address_space(3))) v4i*)(p))

__device__ __forceinline__ unsigned short f2bf(float f) {
  union { float f; unsigned u; } v; v.f = f;
  unsigned r = v.u + 0x7FFFu + ((v.u >> 16) & 1u);
  return (unsigned short)(r >> 16);
}
__device__ __forceinline__ float bf2f(unsigned short h) {
  union { unsigned u; float f; } v; v.u = ((unsigned)h) << 16;
  return v.f;
}

// ---------- problem constants ----------
#define OBS_DIM 128
#define ACT_DIM 32
#define NK      16
#define H_DIM   1024
#define B_DIM   4096
#define IN1     (OBS_DIM + ACT_DIM)   // 160

// ---------- prep: concat obs||act -> bf16 ----------
__global__ __launch_bounds__(256) void prep_concat(
    const float* __restrict__ obs, const float* __restrict__ act,
    unsigned short* __restrict__ xc) {
  int idx = blockIdx.x * 256 + threadIdx.x;  // over B_DIM*IN1
  int b = idx / IN1, c = idx - b * IN1;
  float v = (c < OBS_DIM) ? obs[b * OBS_DIM + c] : act[b * ACT_DIM + (c - OBS_DIM)];
  xc[idx] = f2bf(v);
}

// ---------- prep: W [KTOT, N] fp32 -> Wt [KTOT/32][N][32] bf16 (GEMM-native) ----------
__global__ __launch_bounds__(256) void cvt_w_transpose(
    const float* __restrict__ W, unsigned short* __restrict__ Wt, int N) {
  const int nblks = N >> 8;
  const int chunk = blockIdx.x / nblks;
  const int n     = (blockIdx.x - chunk * nblks) * 256 + threadIdx.x;
  const float* src = W + (long)chunk * 32 * N + n;
  v8u16 o[4];
#pragma unroll
  for (int kk = 0; kk < 32; ++kk)
    o[kk >> 3][kk & 7] = f2bf(src[(long)kk * N]);
  unsigned short* dst = Wt + ((long)chunk * N + n) * 32;
  *(v8u16*)(dst)      = o[0];
  *(v8u16*)(dst + 8)  = o[1];
  *(v8u16*)(dst + 16) = o[2];
  *(v8u16*)(dst + 24) = o[3];
}

// ---------- prep: cw [B,16] -> bf16 padded [B,32] ----------
__global__ __launch_bounds__(256) void pad_cw_kernel(
    const float* __restrict__ cw, unsigned short* __restrict__ cwp) {
  int idx = blockIdx.x * 256 + threadIdx.x;  // over B_DIM*32
  int b = idx >> 5, c = idx & 31;
  cwp[idx] = (c < NK) ? f2bf(cw[b * NK + c]) : (unsigned short)0;
}

// ---------- prep: bias [16,N] -> transposed padded bf16 [N][32] ----------
__global__ __launch_bounds__(256) void pad_bias_t(
    const float* __restrict__ bias, unsigned short* __restrict__ bbT) {
  int idx = blockIdx.x * 256 + threadIdx.x;  // over H_DIM*32
  int n = idx >> 5, kk = idx & 31;
  bbT[idx] = (kk < NK) ? f2bf(bias[kk * H_DIM + n]) : (unsigned short)0;
}

// ---------- compositional GEMM via WMMA bf16 ----------
// accT[b,o] = sum_kseg cw[b,kseg] * ( sum_i A[b,i]*W[kseg*IN_DIM+i, o] )
//           + rank-32 WMMA update with (cw padded) x (bias padded)
// Block 128x128, 8 waves (4Mx2N), wave tile 32x64 (2x4 WMMA), K-step 32.
// Triple-buffered LDS, async global->LDS copies, one barrier per chunk.
template <int IN_DIM>
__global__ __launch_bounds__(256) void comp_gemm(
    const unsigned short* __restrict__ Asrc,  // [B_DIM, IN_DIM] bf16
    const unsigned short* __restrict__ cwp,   // [B_DIM, 32] bf16 (cols 16..31 = 0)
    const unsigned short* __restrict__ Wt,    // [KTOT/32][N][32] bf16
    const unsigned short* __restrict__ bbT,   // [N][32] bf16 (cols 16..31 = 0)
    const float* __restrict__ cw,             // [B_DIM, NK] fp32 (segment fold)
    unsigned short* __restrict__ Out,         // [B_DIM, N] bf16 (ReLU applied)
    int N) {
  constexpr int CPS   = IN_DIM / 32;       // chunks per k-segment
  constexpr int CHW   = NK * CPS;          // weight chunks
  constexpr int CH    = CHW + 1;           // + bias chunk
  constexpr int LDAP  = 40;                // padded LDS row stride: 80B, 16B-aligned

  __shared__ unsigned short ldsA[3][128 * LDAP];   // [m][k]
  __shared__ unsigned short ldsB[3][128 * LDAP];   // [n][k]

  const int t      = threadIdx.x;
  const int blockM = blockIdx.y * 128;
  const int blockN = blockIdx.x * 128;

  const int wave  = t >> 5;
  const int lane  = t & 31;
  const int waveM = (wave >> 1) * 32;
  const int waveN = (wave & 1) * 64;
  const int lm    = lane & 15;
  const int lk    = lane >> 4;

  // staging roles: each thread copies 32B of A and 32B of B per chunk
  const int srow  = t >> 1;            // 0..127
  const int shalf = (t & 1) * 16;      // 0 or 16

  const v8f zero = {};
  v8f accT[2][4] = {};
  v8f accS[2][4] = {};

  auto stageChunk = [&](int c) {
    const int buf = c % 3;
    const unsigned short* ga;
    const unsigned short* gb;
    if (c < CHW) {
      ga = Asrc + (long)(blockM + srow) * IN_DIM + (c % CPS) * 32 + shalf;
      gb = Wt + ((long)c * N + blockN + srow) * 32 + shalf;
    } else {  // bias chunk
      ga = cwp + (long)(blockM + srow) * 32 + shalf;
      gb = bbT + (long)(blockN + srow) * 32 + shalf;
    }
    unsigned short* la = &ldsA[buf][srow * LDAP + shalf];
    unsigned short* lb = &ldsB[buf][srow * LDAP + shalf];
#if USE_ASYNC
    __builtin_amdgcn_global_load_async_to_lds_b128(ASG(ga), ASL(la), 0, 0);
    __builtin_amdgcn_global_load_async_to_lds_b128(ASG(ga + 8), ASL(la + 8), 0, 0);
    __builtin_amdgcn_global_load_async_to_lds_b128(ASG(gb), ASL(lb), 0, 0);
    __builtin_amdgcn_global_load_async_to_lds_b128(ASG(gb + 8), ASL(lb + 8), 0, 0);
#else
    const v8u16* gav = (const v8u16*)ga;
    const v8u16* gbv = (const v8u16*)gb;
    v8u16 a0 = gav[0], a1 = gav[1], b0 = gbv[0], b1 = gbv[1];
    *(v8u16*)la = a0; *(v8u16*)(la + 8) = a1;
    *(v8u16*)lb = b0; *(v8u16*)(lb + 8) = b1;
#endif
  };

  // fragments per documented wave32 layouts + 8 WMMA
  auto compute = [&](int buf, v8f (&acc)[2][4]) {
    v16bf a[2], b[4];
#pragma unroll
    for (int mi = 0; mi < 2; ++mi) {
      const int r = waveM + mi * 16 + lm;
      v8u16 lo = *(const v8u16*)&ldsA[buf][r * LDAP + lk * 8];       // K=lk*8..+7
      v8u16 hi = *(const v8u16*)&ldsA[buf][r * LDAP + 16 + lk * 8];  // K=16+lk*8..+7
      v16u16 f = __builtin_shufflevector(lo, hi, 0, 1, 2, 3, 4, 5, 6, 7,
                                         8, 9, 10, 11, 12, 13, 14, 15);
      a[mi] = __builtin_bit_cast(v16bf, f);
    }
#pragma unroll
    for (int ni = 0; ni < 4; ++ni) {
      const int n = waveN + ni * 16 + lm;
      v8u16 lo = *(const v8u16*)&ldsB[buf][n * LDAP + lk * 8];       // K=lk*8..+7
      v8u16 hi = *(const v8u16*)&ldsB[buf][n * LDAP + 16 + lk * 8];  // K=16+lk*8..+7
      v16u16 f = __builtin_shufflevector(lo, hi, 0, 1, 2, 3, 4, 5, 6, 7,
                                         8, 9, 10, 11, 12, 13, 14, 15);
      b[ni] = __builtin_bit_cast(v16bf, f);
    }
#pragma unroll
    for (int mi = 0; mi < 2; ++mi)
#pragma unroll
      for (int ni = 0; ni < 4; ++ni)
        acc[mi][ni] = __builtin_amdgcn_wmma_f32_16x16x32_bf16(
            false, a[mi], false, b[ni], (short)0, acc[mi][ni], false, false);
  };

  stageChunk(0);
  for (int c = 0; c < CH; ++c) {
    if (c + 1 < CH) {
      stageChunk(c + 1);
#if USE_ASYNC
      __builtin_amdgcn_s_wait_asynccnt(4);  // in-order: chunk c's 4 copies done
#endif
    } else {
#if USE_ASYNC
      __builtin_amdgcn_s_wait_asynccnt(0);
#endif
    }
    __syncthreads();

    if (c < CHW) {
      compute(c % 3, accS);
      if ((c % CPS) == CPS - 1) {
        const int kseg = c / CPS;
#pragma unroll
        for (int mi = 0; mi < 2; ++mi) {
#pragma unroll
          for (int r = 0; r < 8; ++r) {
            const int row = blockM + waveM + mi * 16 + lk * 8 + r;
            const float s = cw[row * NK + kseg];
#pragma unroll
            for (int ni = 0; ni < 4; ++ni) {
              accT[mi][ni][r] += s * accS[mi][ni][r];
            }
          }
        }
#pragma unroll
        for (int mi = 0; mi < 2; ++mi)
#pragma unroll
          for (int ni = 0; ni < 4; ++ni) accS[mi][ni] = zero;
      }
    } else {
      compute(c % 3, accT);  // composed bias, weight 1
    }
  }

  // ---- ReLU + store bf16
#pragma unroll
  for (int mi = 0; mi < 2; ++mi)
#pragma unroll
    for (int ni = 0; ni < 4; ++ni) {
      const int col = blockN + waveN + ni * 16 + lm;
#pragma unroll
      for (int r = 0; r < 8; ++r) {
        const int row = blockM + waveM + mi * 16 + lk * 8 + r;
        float v = accT[mi][ni][r];
        v = v > 0.f ? v : 0.f;
        Out[(long)row * N + col] = f2bf(v);
      }
    }
}

// ---------- out[b] = h2[b,:] . Wo + bo ----------
__global__ __launch_bounds__(256) void out_proj(
    const unsigned short* __restrict__ h2, const float* __restrict__ Wo,
    const float* __restrict__ bo, float* __restrict__ out) {
  const int wave = threadIdx.x >> 5;
  const int lane = threadIdx.x & 31;
  const int row  = blockIdx.x * 8 + wave;
  float s = 0.f;
  for (int o = lane; o < H_DIM; o += 32)
    s += bf2f(h2[(long)row * H_DIM + o]) * Wo[o];
#pragma unroll
  for (int off = 16; off > 0; off >>= 1) s += __shfl_down(s, off, 32);
  if (lane == 0) out[row] = s + bo[0];
}

// ---------- launcher ----------
extern "C" void kernel_launch(void* const* d_in, const int* in_sizes, int n_in,
                              void* d_out, int out_size, void* d_ws, size_t ws_size,
                              hipStream_t stream) {
  (void)in_sizes; (void)n_in; (void)out_size; (void)ws_size;
  const float* obs = (const float*)d_in[0];   // [B, 128]
  const float* act = (const float*)d_in[1];   // [B, 32]
  const float* cwf = (const float*)d_in[2];   // [B, 16]
  const float* W1  = (const float*)d_in[3];   // [16,160,1024]
  const float* b1  = (const float*)d_in[4];   // [16,1024]
  const float* W2  = (const float*)d_in[5];   // [16,1024,1024]
  const float* b2  = (const float*)d_in[6];   // [16,1024]
  const float* Wo  = (const float*)d_in[7];   // [1024,1]
  const float* bo  = (const float*)d_in[8];   // [1]
  float* out = (float*)d_out;                 // [B,1]

  unsigned short* ws  = (unsigned short*)d_ws;
  unsigned short* xc  = ws;                                        // B*160
  unsigned short* h1  = xc  + (size_t)B_DIM * IN1;                 // B*1024
  unsigned short* h2  = h1  + (size_t)B_DIM * H_DIM;               // B*1024
  unsigned short* w1t = h2  + (size_t)B_DIM * H_DIM;               // 80*1024*32
  unsigned short* w2t = w1t + (size_t)NK * IN1 * H_DIM;            // 512*1024*32
  unsigned short* cwp = w2t + (size_t)NK * H_DIM * H_DIM;          // B*32
  unsigned short* b1t = cwp + (size_t)B_DIM * 32;                  // 1024*32
  unsigned short* b2t = b1t + (size_t)32 * H_DIM;                  // 1024*32

  // one-time per-launch bf16 conversions into GEMM-native (transposed) layout
  cvt_w_transpose<<<(NK * IN1 / 32) * (H_DIM / 256), 256, 0, stream>>>(W1, w1t, H_DIM);
  cvt_w_transpose<<<(NK * H_DIM / 32) * (H_DIM / 256), 256, 0, stream>>>(W2, w2t, H_DIM);
  pad_cw_kernel<<<(B_DIM * 32) / 256, 256, 0, stream>>>(cwf, cwp);
  pad_bias_t<<<(H_DIM * 32) / 256, 256, 0, stream>>>(b1, b1t);
  pad_bias_t<<<(H_DIM * 32) / 256, 256, 0, stream>>>(b2, b2t);
  prep_concat<<<(B_DIM * IN1) / 256, 256, 0, stream>>>(obs, act, xc);

  dim3 grid(H_DIM / 128, B_DIM / 128);  // (8, 32)
  comp_gemm<IN1><<<grid, 256, 0, stream>>>(xc, cwp, w1t, b1t, cwf, h1, H_DIM);
  comp_gemm<H_DIM><<<grid, 256, 0, stream>>>(h1, cwp, w2t, b2t, cwf, h2, H_DIM);

  out_proj<<<B_DIM / 8, 256, 0, stream>>>(h2, Wo, bo, out);
}
